// Attention_26182120636694
// MI455X (gfx1250) — matrix operand order repaired and checked by reading, not compiled, not run on hardware
//
#include <hip/hip_runtime.h>
#include <hip/hip_bf16.h>
#include <stdint.h>

typedef __bf16 bf16_t;
typedef __attribute__((ext_vector_type(16))) __bf16 v16bf;
typedef __attribute__((ext_vector_type(8)))  float  v8f;
typedef int v4i __attribute__((vector_size(16)));   // matches async-LDS builtin param

#define ASTR 34   // converting-stage LDS stride (bf16): 17 banks, coprime with 64
#define ASTR3 40  // async-copied A-tile stride (bf16): 80B rows (16B aligned)
#define KSTR 72   // attention K/V tile stride (bf16): 144B rows (16B aligned)
#define PSTR 36   // P tile stride (bf16)

#if __has_builtin(__builtin_amdgcn_global_load_async_to_lds_b128) && \
    __has_builtin(__builtin_amdgcn_s_wait_asynccnt)
#define USE_ASYNC_LDS 1
#else
#define USE_ASYNC_LDS 0
#endif

// 16-byte global -> LDS copy; async (ASYNCcnt-tracked) on gfx1250, synchronous
// VGPR-mediated fallback otherwise.
__device__ __forceinline__ void copy16_g2l(const bf16_t* g, bf16_t* l) {
#if USE_ASYNC_LDS
  __builtin_amdgcn_global_load_async_to_lds_b128(
      (__attribute__((address_space(1))) v4i*)g,
      (__attribute__((address_space(3))) v4i*)l, 0, 0);
#else
  *(uint4*)l = *(const uint4*)g;
#endif
}
__device__ __forceinline__ void async_join() {
#if USE_ASYNC_LDS
  __builtin_amdgcn_s_wait_asynccnt(0);
#endif
}

__device__ __forceinline__ v8f wmma_bf16(v16bf a, v16bf b, v8f c) {
  return __builtin_amdgcn_wmma_f32_16x16x32_bf16(false, a, false, b, (short)0, c,
                                                 false, false);
}

// A fragment (16x32 bf16) from LDS tile a[16][stride].
// lane m=l&15; lanes<16 hold K=0..15, lanes>=16 hold K=16..31.
__device__ __forceinline__ v16bf lds_a_frag(const bf16_t* a, int stride) {
  int lane = threadIdx.x & 31;
  const bf16_t* row = a + (lane & 15) * stride + ((lane < 16) ? 0 : 16);
  v16bf f;
#pragma unroll
  for (int j = 0; j < 8; ++j) f[j] = row[j];
#pragma unroll
  for (int j = 0; j < 8; ++j) f[8 + j] = row[8 + j];
  return f;
}

// B fragment (32x16) from an "n-major" LDS tile bt[n][k] (n = output column).
// lane n=l&15; lanes<16 hold K=0..15, lanes>=16 hold K=16..31.
__device__ __forceinline__ v16bf lds_b_frag(const bf16_t* bt, int stride) {
  int lane = threadIdx.x & 31;
  const bf16_t* row = bt + (lane & 15) * stride + ((lane < 16) ? 0 : 16);
  v16bf f;
#pragma unroll
  for (int j = 0; j < 16; ++j) f[j] = row[j];
  return f;
}

// B fragment (32x16) from a "k-major" LDS tile v[k][n] (used for P@V).
__device__ __forceinline__ v16bf lds_bt_frag(const bf16_t* v, int stride, int nbase) {
  int lane = threadIdx.x & 31;
  int n = nbase + (lane & 15);
  int kb = (lane < 16) ? 0 : 16;
  v16bf f;
#pragma unroll
  for (int j = 0; j < 16; ++j) f[j] = v[(kb + j) * stride + n];
  return f;
}

// ---------------------------------------------------------------------------
// Kernel 1: QKV projection. Y[4096,3072] = X[4096,1024] @ W^T, written as bf16
// into Q/K/V buffers laid out [B=4][H=16][N=1024][D=64].
// Workgroup tile 128x128, 8 waves, 32x64 per wave (8 WMMAs per K-chunk).
// ---------------------------------------------------------------------------
__global__ __launch_bounds__(256) void qkv_gemm(const float* __restrict__ X,
                                                const float* __restrict__ W,
                                                bf16_t* __restrict__ qkv) {
  __shared__ __align__(16) bf16_t Als[128 * ASTR];
  __shared__ __align__(16) bf16_t Bls[128 * ASTR];
  const int t = threadIdx.x;
  const int lane = t & 31;
  const int wid = t >> 5;
  const int wm = wid & 3, wn = wid >> 2;          // 4x2 wave grid
  const int m0 = blockIdx.y * 128;
  const int n0 = blockIdx.x * 128;

  v8f acc[2][4] = {};
  for (int k0 = 0; k0 < 1024; k0 += 32) {
    __syncthreads();
    {
      int r = t >> 3, c4 = t & 7;
#pragma unroll
      for (int s = 0; s < 4; ++s) {               // A: 128x32 fp32 -> bf16
        int row = r + 32 * s;
        float4 v = *(const float4*)(X + (size_t)(m0 + row) * 1024 + k0 + c4 * 4);
        bf16_t* dst = Als + row * ASTR + c4 * 4;
        dst[0] = (bf16_t)v.x; dst[1] = (bf16_t)v.y;
        dst[2] = (bf16_t)v.z; dst[3] = (bf16_t)v.w;
      }
#pragma unroll
      for (int s = 0; s < 4; ++s) {               // B: 128 rows of W, 32 cols
        int row = r + 32 * s;
        float4 v = *(const float4*)(W + (size_t)(n0 + row) * 1024 + k0 + c4 * 4);
        bf16_t* dst = Bls + row * ASTR + c4 * 4;
        dst[0] = (bf16_t)v.x; dst[1] = (bf16_t)v.y;
        dst[2] = (bf16_t)v.z; dst[3] = (bf16_t)v.w;
      }
    }
    __syncthreads();
    v16bf a0 = lds_a_frag(Als + (wm * 32) * ASTR, ASTR);
    v16bf a1 = lds_a_frag(Als + (wm * 32 + 16) * ASTR, ASTR);
#pragma unroll
    for (int in = 0; in < 4; ++in) {
      v16bf b = lds_b_frag(Bls + (wn * 64 + in * 16) * ASTR, ASTR);
      acc[0][in] = wmma_bf16(a0, b, acc[0][in]);
      acc[1][in] = wmma_bf16(a1, b, acc[1][in]);
    }
  }
  const int half8 = (lane < 16) ? 0 : 8;
  const int ncol = lane & 15;
#pragma unroll
  for (int im = 0; im < 2; ++im)
#pragma unroll
    for (int in = 0; in < 4; ++in)
#pragma unroll
      for (int i = 0; i < 8; ++i) {
        int row = m0 + wm * 32 + im * 16 + i + half8;   // token row in [0,4096)
        int n = n0 + wn * 64 + in * 16 + ncol;          // [0,3072)
        int which = n >> 10, h = (n >> 6) & 15, d = n & 63;
        int b = row >> 10, tok = row & 1023;
        size_t off = (size_t)which * (4u * 16 * 1024 * 64) +
                     (((size_t)(b * 16 + h) * 1024 + tok) * 64 + d);
        qkv[off] = (bf16_t)acc[im][in][i];
      }
}

// ---------------------------------------------------------------------------
// Kernel 2: flash attention (online softmax), double-buffered async K/V tiles.
// Block = 4 waves; each wave owns a 16-row query tile; waves share LDS K/V
// chunks (32 keys) for one (branch,b,h). Async copy of chunk i+1 overlaps the
// WMMAs + softmax of chunk i. Output bf16 [2][B][N][C], x_v half first.
// ---------------------------------------------------------------------------
__global__ __launch_bounds__(128) void flash_attn(const bf16_t* __restrict__ Qb,
                                                  const bf16_t* __restrict__ Kb,
                                                  const bf16_t* __restrict__ Vb,
                                                  const float* __restrict__ mask,
                                                  bf16_t* __restrict__ attn) {
  __shared__ __align__(16) bf16_t Kls[2][32 * KSTR];
  __shared__ __align__(16) bf16_t Vls[2][32 * KSTR];
  __shared__ bf16_t Pls[4][16 * PSTR];

  const int t = threadIdx.x;
  const int lane = t & 31;
  const int wid = t >> 5;

  const int qgroup = blockIdx.x & 15;
  const int bh = (blockIdx.x >> 4) & 63;
  const int br = blockIdx.x >> 10;     // 0 = (q,k,v) branch, 1 = (v,v,v) branch
  const int q0 = qgroup * 64 + wid * 16;

  const bf16_t* Qp = br ? Vb : Qb;
  const bf16_t* Kp = br ? Vb : Kb;
  const bf16_t* Vp = Vb;
  const float* mb = mask + (size_t)br * 1024 * 1024;
  const size_t kvbase = (size_t)bh * 1024;

  // Q fragments (16x64 tile as two K=32 A-fragments), loaded once from global.
  v16bf aq[2];
  {
    const bf16_t* qbase = Qp + (kvbase + q0) * 64;
    int m = lane & 15;
    int kb = (lane < 16) ? 0 : 16;
#pragma unroll
    for (int s = 0; s < 2; ++s) {
      const bf16_t* p = qbase + m * 64 + s * 32;
#pragma unroll
      for (int j = 0; j < 8; ++j) aq[s][j] = p[kb + j];
#pragma unroll
      for (int j = 0; j < 8; ++j) aq[s][8 + j] = p[kb + 8 + j];
    }
  }

  v8f o[4] = {};
  float mrun[8], lrun[8];
#pragma unroll
  for (int i = 0; i < 8; ++i) { mrun[i] = -3.0e38f; lrun[i] = 0.f; }

  const float scale = 0.125f;          // 1/sqrt(D=64)
  const int half8 = (lane < 16) ? 0 : 8;
  const int ncol = lane & 15;

  // Stage chunk 0 into buffer 0.
#pragma unroll
  for (int rep = 0; rep < 2; ++rep) {
    int slot = t + rep * 128;
    int row = slot >> 3, seg = slot & 7;
    size_t go = (kvbase + row) * 64 + seg * 8;
    int lo = row * KSTR + seg * 8;
    copy16_g2l(Kp + go, Kls[0] + lo);
    copy16_g2l(Vp + go, Vls[0] + lo);
  }
  async_join();
  __syncthreads();

  for (int ic = 0; ic < 32; ++ic) {
    const int kc = ic * 32;
    const bf16_t* Kc = Kls[ic & 1];
    const bf16_t* Vc = Vls[ic & 1];

    if (ic + 1 < 32) {                 // prefetch next chunk into other buffer
      bf16_t* Kd = Kls[(ic + 1) & 1];
      bf16_t* Vd = Vls[(ic + 1) & 1];
#pragma unroll
      for (int rep = 0; rep < 2; ++rep) {
        int slot = t + rep * 128;
        int row = slot >> 3, seg = slot & 7;
        size_t go = (kvbase + kc + 32 + row) * 64 + seg * 8;
        int lo = row * KSTR + seg * 8;
        copy16_g2l(Kp + go, Kd + lo);
        copy16_g2l(Vp + go, Vd + lo);
      }
    }

    // S tile 16x32: two 16x16 f32 accumulators, K=64 in two bf16 WMMA steps.
    v8f c0 = {}, c1 = {};
    c0 = wmma_bf16(aq[0], lds_b_frag(Kc, KSTR), c0);
    c0 = wmma_bf16(aq[1], lds_b_frag(Kc + 32, KSTR), c0);
    c1 = wmma_bf16(aq[0], lds_b_frag(Kc + 16 * KSTR, KSTR), c1);
    c1 = wmma_bf16(aq[1], lds_b_frag(Kc + 16 * KSTR + 32, KSTR), c1);

    bf16_t* Pw = Pls[wid];
    float alpha[8];
#pragma unroll
    for (int i = 0; i < 8; ++i) {
      int row = q0 + i + half8;
      float s0 = c0[i] * scale + mb[(size_t)row * 1024 + kc + ncol];
      float s1 = c1[i] * scale + mb[(size_t)row * 1024 + kc + 16 + ncol];
      float mx = fmaxf(s0, s1);
#pragma unroll
      for (int off = 1; off < 16; off <<= 1) mx = fmaxf(mx, __shfl_xor(mx, off));
      float mn = fmaxf(mrun[i], mx);
      float a = __expf(mrun[i] - mn);
      float p0 = __expf(s0 - mn);
      float p1 = __expf(s1 - mn);
      float rs = p0 + p1;
#pragma unroll
      for (int off = 1; off < 16; off <<= 1) rs += __shfl_xor(rs, off);
      lrun[i] = lrun[i] * a + rs;
      mrun[i] = mn;
      alpha[i] = a;
      Pw[(i + half8) * PSTR + ncol] = (bf16_t)p0;       // C-layout -> LDS
      Pw[(i + half8) * PSTR + 16 + ncol] = (bf16_t)p1;
    }
#pragma unroll
    for (int f = 0; f < 4; ++f)
#pragma unroll
      for (int i = 0; i < 8; ++i) o[f][i] *= alpha[i];

    v16bf ap = lds_a_frag(Pw, PSTR);                    // LDS -> A-layout
#pragma unroll
    for (int f = 0; f < 4; ++f)
      o[f] = wmma_bf16(ap, lds_bt_frag(Vc, KSTR, f * 16), o[f]);

    async_join();          // my prefetch copies done
    __syncthreads();       // everyone's copies visible; safe to swap buffers
  }

  // Write: branch 1 (x_v) -> first half, branch 0 (x_ori) -> second half.
  int b = bh >> 4, h = bh & 15;
  size_t base = (size_t)(br ? 0 : 1) * 4096 * 1024;
#pragma unroll
  for (int f = 0; f < 4; ++f)
#pragma unroll
    for (int i = 0; i < 8; ++i) {
      int tok = q0 + i + half8;
      int d = f * 16 + ncol;
      attn[base + ((size_t)b * 1024 + tok) * 1024 + h * 64 + d] =
          (bf16_t)(o[f][i] / lrun[i]);
    }
}

// ---------------------------------------------------------------------------
// Kernel 3: output projection. out[8192,1024] (fp32) =
// attn[8192,1024](bf16) @ proj_w^T + proj_b. A tile staged with async copies.
// Workgroup tile 128x128, 32x64 per wave.
// ---------------------------------------------------------------------------
__global__ __launch_bounds__(256) void proj_gemm(const bf16_t* __restrict__ A,
                                                 const float* __restrict__ W,
                                                 const float* __restrict__ bias,
                                                 float* __restrict__ out) {
  __shared__ __align__(16) bf16_t Als[128 * ASTR3];
  __shared__ __align__(16) bf16_t Bls[128 * ASTR];
  const int t = threadIdx.x;
  const int lane = t & 31;
  const int wid = t >> 5;
  const int wm = wid & 3, wn = wid >> 2;
  const int m0 = blockIdx.y * 128;
  const int n0 = blockIdx.x * 128;

  v8f acc[2][4] = {};
  for (int k0 = 0; k0 < 1024; k0 += 32) {
    __syncthreads();
#pragma unroll
    for (int rep = 0; rep < 2; ++rep) {            // A: 128x32 bf16, async copy
      int slot = t + rep * 256;
      int row = slot >> 2, seg = slot & 3;
      copy16_g2l(A + (size_t)(m0 + row) * 1024 + k0 + seg * 8,
                 Als + row * ASTR3 + seg * 8);
    }
    {
      int r = t >> 3, c4 = t & 7;
#pragma unroll
      for (int s = 0; s < 4; ++s) {                // B: 128x32 fp32 -> bf16
        int row = r + 32 * s;
        float4 v = *(const float4*)(W + (size_t)(n0 + row) * 1024 + k0 + c4 * 4);
        bf16_t* dst = Bls + row * ASTR + c4 * 4;
        dst[0] = (bf16_t)v.x; dst[1] = (bf16_t)v.y;
        dst[2] = (bf16_t)v.z; dst[3] = (bf16_t)v.w;
      }
    }
    async_join();
    __syncthreads();
    v16bf a0 = lds_a_frag(Als + (wm * 32) * ASTR3, ASTR3);
    v16bf a1 = lds_a_frag(Als + (wm * 32 + 16) * ASTR3, ASTR3);
#pragma unroll
    for (int in = 0; in < 4; ++in) {
      v16bf b = lds_b_frag(Bls + (wn * 64 + in * 16) * ASTR, ASTR);
      acc[0][in] = wmma_bf16(a0, b, acc[0][in]);
      acc[1][in] = wmma_bf16(a1, b, acc[1][in]);
    }
  }
  const int half8 = (lane < 16) ? 0 : 8;
  const int ncol = lane & 15;
#pragma unroll
  for (int im = 0; im < 2; ++im)
#pragma unroll
    for (int in = 0; in < 4; ++in) {
      int n = n0 + wn * 64 + in * 16 + ncol;
      float bv = bias[n];
#pragma unroll
      for (int i = 0; i < 8; ++i) {
        int row = m0 + wm * 32 + im * 16 + i + half8;
        out[(size_t)row * 1024 + n] = acc[im][in][i] + bv;
      }
    }
}

extern "C" void kernel_launch(void* const* d_in, const int* in_sizes, int n_in,
                              void* d_out, int out_size, void* d_ws, size_t ws_size,
                              hipStream_t stream) {
  (void)in_sizes; (void)n_in; (void)out_size; (void)ws_size;
  const float* x      = (const float*)d_in[0];
  const float* mask   = (const float*)d_in[1];
  const float* qkv_w  = (const float*)d_in[2];
  const float* proj_w = (const float*)d_in[3];
  const float* proj_b = (const float*)d_in[4];
  float* out = (float*)d_out;

  // Workspace layout (bf16): Q | K | V (4M elems each, [B][H][N][D]) then
  // attn (8M elems, [2][B][N][C]) -> 40 MB total.
  bf16_t* Qb   = (bf16_t*)d_ws;
  bf16_t* Kb   = Qb + (size_t)4 * 1024 * 1024;
  bf16_t* Vb   = Qb + (size_t)8 * 1024 * 1024;
  bf16_t* attn = Qb + (size_t)12 * 1024 * 1024;

  qkv_gemm<<<dim3(24, 32), 256, 0, stream>>>(x, qkv_w, Qb);
  flash_attn<<<dim3(2048), 128, 0, stream>>>(Qb, Kb, Vb, mask, attn);
  proj_gemm<<<dim3(8, 64), 256, 0, stream>>>(attn, proj_w, proj_b, out);
}
